// VariLengthInputLayer_17265768530663
// MI455X (gfx1250) — matrix operand order repaired and checked by reading, not compiled, not run on hardware
//
#include <hip/hip_runtime.h>
#include <hip/hip_bf16.h>
#include <math.h>

typedef _Float16 half2v  __attribute__((ext_vector_type(2)));
typedef _Float16 half4v  __attribute__((ext_vector_type(4)));
typedef _Float16 half8   __attribute__((ext_vector_type(8)));
typedef _Float16 half16  __attribute__((ext_vector_type(16)));
typedef float    float4v __attribute__((ext_vector_type(4)));
typedef float    float8  __attribute__((ext_vector_type(8)));

#define BS    16384
#define HEADS 8
#define DM    256
#define NQKV  768
#define TEMP_INV (1.0f/5.656854249492381f)

// ---------------- WMMA fragment loaders (CDNA5 16x16x32 f16 layouts) ----------

// A: 16x32 f16. lanes 0-15: M=lane, halves0..7 = K0..7, halves8..15 = K16..23
//               lanes16-31: M=lane-16, halves0..7 = K8..15, halves8..15 = K24..31
__device__ inline half16 load_a_frag(const _Float16* lA, int lane, int rowBase, int ldah) {
    const int g  = lane >> 4;
    const int ml = lane & 15;
    const _Float16* p = lA + (size_t)(rowBase + ml) * ldah;
    half8 lo = *(const half8*)(p + g * 8);
    half8 hi = *(const half8*)(p + 16 + g * 8);
    return __builtin_shufflevector(lo, hi, 0,1,2,3,4,5,6,7,8,9,10,11,12,13,14,15);
}

// B: 32x16 f16 stored transposed in LDS as [n][k].
// lanes 0-15: N=lane, halves0..7 = K0..7, halves8..15 = K16..23 ; lanes16-31: N=lane-16, halves0..7 = K8..15, halves8..15 = K24..31
__device__ inline half16 load_b_frag(const _Float16* lBt, int lane, int colBase, int ldbh) {
    const int g  = lane >> 4;
    const int nl = lane & 15;
    const _Float16* p = lBt + (size_t)(colBase + nl) * ldbh;
    half8 lo = *(const half8*)(p + g * 8);
    half8 hi = *(const half8*)(p + 16 + g * 8);
    return __builtin_shufflevector(lo, hi, 0,1,2,3,4,5,6,7,8,9,10,11,12,13,14,15);
}

__device__ inline float8 wmma32(half16 a, half16 b, float8 c) {
    return __builtin_amdgcn_wmma_f32_16x16x32_f16(false, a, false, b, (short)0, c, false, false);
}

// ---------------- Kernel 0: init global-min cell --------------------------------
__global__ void init_gmin_kernel(unsigned* gmin) {
    if (threadIdx.x == 0 && blockIdx.x == 0) *gmin = 0xFFFFFFFFu;
}

// ---------------- Kernel 1: Q/K/V projection GEMMs ------------------------------
struct ProjParams {
    const float* X;
    const float* W[9];   // [seg*3 + m], seg: 0=q 1=k 2=v
    const float* Bi[9];
    float*       outbase;
};

#define P_BM 128
#define P_BN 128
#define P_BK 32
#define P_LDH 40   // halves per LDS row (32 + 8 pad) = 80B, 16B aligned

// Stage next A tile (128x32 fp32) into registers. 1024 float4 groups, 4/thread.
__device__ inline void gload_A(const float* __restrict__ X, int bm0, int xoff,
                               int d, int k0, bool tail, int tid, float4v ra[4]) {
    #pragma unroll
    for (int j = 0; j < 4; ++j) {
        int grp = j * 256 + tid;
        int r   = grp >> 3;
        int c4  = (grp & 7) * 4;
        const float* base = X + (size_t)(bm0 + r) * 2900 + xoff + k0 + c4;
        if (!tail) {
            ra[j] = *(const float4v*)base;          // 16B-aligned
        } else {
            int rem = d - k0;
            #pragma unroll
            for (int e = 0; e < 4; ++e) ra[j][e] = (c4 + e < rem) ? base[e] : 0.0f;
        }
    }
}

__device__ inline void store_A(_Float16* lA, int tid, const float4v ra[4]) {
    #pragma unroll
    for (int j = 0; j < 4; ++j) {
        int grp = j * 256 + tid;
        int r   = grp >> 3;
        int c4  = (grp & 7) * 4;
        half4v h = { (_Float16)ra[j][0], (_Float16)ra[j][1],
                     (_Float16)ra[j][2], (_Float16)ra[j][3] };
        *(half4v*)(lA + r * P_LDH + c4) = h;        // 8B-aligned ds_store_b64
    }
}

// Stage next B tile (32x128 fp32) as k-row pairs. 512 items, 2/thread.
__device__ inline void gload_B(const float* __restrict__ Wsel, int ncc0, int d,
                               int k0, bool tail, int tid, float4v rb[2][2]) {
    float4v z = {0.0f, 0.0f, 0.0f, 0.0f};
    #pragma unroll
    for (int it = 0; it < 2; ++it) {
        int item = it * 256 + tid;
        int kk   = (item >> 5) * 2;                 // 0,2,..,30
        int n4   = (item & 31) * 4;                 // 0,4,..,124
        const float* base = Wsel + (size_t)(k0 + kk) * DM + ncc0 + n4;
        if (!tail) {
            rb[it][0] = *(const float4v*)base;
            rb[it][1] = *(const float4v*)(base + DM);
        } else {
            int rem = d - k0;
            rb[it][0] = (kk     < rem) ? *(const float4v*)base        : z;
            rb[it][1] = (kk + 1 < rem) ? *(const float4v*)(base + DM) : z;
        }
    }
}

__device__ inline void store_B(_Float16* lB, int tid, const float4v rb[2][2]) {
    #pragma unroll
    for (int it = 0; it < 2; ++it) {
        int item = it * 256 + tid;
        int kk   = (item >> 5) * 2;
        int n4   = (item & 31) * 4;
        #pragma unroll
        for (int i = 0; i < 4; ++i) {
            half2v h = { (_Float16)rb[it][0][i], (_Float16)rb[it][1][i] };
            *(half2v*)(lB + (n4 + i) * P_LDH + kk) = h;   // contiguous k-pair, b32
        }
    }
}

__global__ __launch_bounds__(256) void proj_gemm(ProjParams p) {
    __shared__ _Float16 lA[2][P_BM * P_LDH];
    __shared__ _Float16 lB[2][P_BN * P_LDH];

    const int dims[3]  = {2000, 500, 400};
    const int xoffs[3] = {0, 2000, 2500};

    const int m    = blockIdx.z;
    const int d    = dims[m];
    const int xoff = xoffs[m];
    const int bm0  = blockIdx.x * P_BM;
    const int n0   = blockIdx.y * P_BN;
    const int seg  = n0 >> 8;
    const int ncc0 = n0 & 255;

    const float* __restrict__ Wsel = p.W[seg * 3 + m];
    const float* __restrict__ Bsel = p.Bi[seg * 3 + m];
    float* __restrict__ outp = p.outbase + (size_t)seg * BS * NQKV;

    const int tid  = threadIdx.x;
    const int wave = tid >> 5;
    const int lane = tid & 31;
    const int wm   = wave & 1;
    const int wn   = wave >> 1;

    float8 acc[4][2] = {};
    float4v ra[4];
    float4v rb[2][2];

    const int ksteps = (d + P_BK - 1) / P_BK;

    // prologue: stage tile 0
    gload_A(p.X, bm0, xoff, d, 0, (ksteps == 1), tid, ra);
    gload_B(Wsel, ncc0, d, 0, (ksteps == 1), tid, rb);

    for (int ks = 0; ks < ksteps; ++ks) {
        const int cur = ks & 1;
        _Float16* cA = lA[cur];
        _Float16* cB = lB[cur];
        store_A(cA, tid, ra);
        store_B(cB, tid, rb);
        __syncthreads();

        // issue next tile's global loads before the math (latency overlap)
        if (ks + 1 < ksteps) {
            const int kn = (ks + 1) * P_BK;
            const bool tail = (ks + 2 == ksteps + 0) && true ? (ks + 1 == ksteps - 1) : false;
            gload_A(p.X, bm0, xoff, d, kn, (ks + 1 == ksteps - 1), tid, ra);
            gload_B(Wsel, ncc0, d, kn, (ks + 1 == ksteps - 1), tid, rb);
            (void)tail;
        }

        half16 bf[2];
        #pragma unroll
        for (int nt = 0; nt < 2; ++nt)
            bf[nt] = load_b_frag(cB, lane, wn * 32 + nt * 16, P_LDH);
        #pragma unroll
        for (int mt = 0; mt < 4; ++mt) {
            half16 af = load_a_frag(cA, lane, wm * 64 + mt * 16, P_LDH);
            #pragma unroll
            for (int nt = 0; nt < 2; ++nt)
                acc[mt][nt] = wmma32(af, bf[nt], acc[mt][nt]);
        }
    }

    // Epilogue: D layout VGPR i -> M = i + 8*(lane>=16), N = lane%16
    const int g  = lane >> 4;
    const int nl = lane & 15;
    #pragma unroll
    for (int mt = 0; mt < 4; ++mt) {
        #pragma unroll
        for (int nt = 0; nt < 2; ++nt) {
            int cc = ncc0 + wn * 32 + nt * 16 + nl;
            float bias = Bsel[cc];
            #pragma unroll
            for (int i = 0; i < 8; ++i) {
                int row = bm0 + wm * 64 + mt * 16 + g * 8 + i;
                outp[(size_t)row * NQKV + m * DM + cc] = acc[mt][nt][i] + bias;
            }
        }
    }
}

// ---------------- Kernel 2: attention scores + global min ----------------------
__global__ __launch_bounds__(256) void attn_scores(const float* __restrict__ qb,
                                                   const float* __restrict__ kb,
                                                   float* __restrict__ attn,
                                                   unsigned* __restrict__ gmin) {
    const int t = blockIdx.x * 256 + threadIdx.x;   // BS*72 exactly
    int b  = t / 72;
    int r  = t % 72;
    int h  = r / 9;
    int mi = (r / 3) % 3;
    int ni = r % 3;
    const float4v* q4 = (const float4v*)(qb + (size_t)b * NQKV + mi * DM + h * 32);
    const float4v* k4 = (const float4v*)(kb + (size_t)b * NQKV + ni * DM + h * 32);
    float acc = 0.0f;
    #pragma unroll
    for (int i = 0; i < 8; ++i) {
        float4v qv = q4[i], kv = k4[i];
        acc += qv[0]*kv[0] + qv[1]*kv[1] + qv[2]*kv[2] + qv[3]*kv[3];
    }
    acc *= TEMP_INV;
    attn[t] = acc;

    __shared__ float red[256];
    red[threadIdx.x] = acc;
    __syncthreads();
    for (int off = 128; off > 0; off >>= 1) {
        if (threadIdx.x < off) red[threadIdx.x] = fminf(red[threadIdx.x], red[threadIdx.x + off]);
        __syncthreads();
    }
    if (threadIdx.x == 0) {
        unsigned bits = __float_as_uint(red[0]);
        unsigned key  = (bits & 0x80000000u) ? ~bits : (bits | 0x80000000u);
        atomicMin(gmin, key);
    }
}

// ---------------- Kernel 3: softmax + o, FC WMMA GEMM, LN, output head ---------
#define T_SPB  16
#define T_ROWS 48
#define T_LDOA 264           // halves, 528B row
#define T_LDBT 40            // halves
#define T_LDX  260           // floats

__global__ __launch_bounds__(256) void fused_tail(const float* __restrict__ attn,
                                                  const float* __restrict__ vbuf,
                                                  const unsigned* __restrict__ gmin,
                                                  const float* __restrict__ fcw,
                                                  const float* __restrict__ fcb,
                                                  const float* __restrict__ lng,
                                                  const float* __restrict__ lnb,
                                                  const float* __restrict__ outw,
                                                  const float* __restrict__ outb,
                                                  float* __restrict__ out) {
    __shared__ __align__(16) unsigned char smem[49920];
    _Float16* oA = (_Float16*)smem;                    // [48][264] halves
    _Float16* bT = (_Float16*)(smem + 25344);          // [256][40] halves
    float*    xs = (float*)smem;                       // [48][260] floats (aliases after GEMM)

    const int b0   = blockIdx.x * T_SPB;
    const int tid  = threadIdx.x;
    const int wave = tid >> 5;
    const int lane = tid & 31;

    unsigned key  = *gmin;
    unsigned bits = (key & 0x80000000u) ? (key ^ 0x80000000u) : ~key;
    float ginv = 1.0f / fabsf(__uint_as_float(bits));

    // Phase 1: scale / L1 / softmax / o = attn @ v  -> f16 tile oA
    for (int u = tid; u < T_SPB * HEADS * 3; u += 256) {
        int si = u / 24, rem = u % 24;
        int h  = rem / 3, mi = rem % 3;
        int b  = b0 + si;
        const float* ab = attn + (size_t)b * 72 + h * 9 + mi * 3;
        float a0 = ab[0] * ginv, a1 = ab[1] * ginv, a2 = ab[2] * ginv;
        float l1 = fmaxf(fabsf(a0) + fabsf(a1) + fabsf(a2), 1e-12f);
        a0 /= l1; a1 /= l1; a2 /= l1;
        float mx = fmaxf(a0, fmaxf(a1, a2));
        float e0 = __expf(a0 - mx), e1 = __expf(a1 - mx), e2 = __expf(a2 - mx);
        float inv = 1.0f / (e0 + e1 + e2);
        float p0 = e0 * inv, p1 = e1 * inv, p2 = e2 * inv;
        const float* vb = vbuf + (size_t)b * NQKV;
        int row = si * 3 + mi;
        #pragma unroll
        for (int dd = 0; dd < 32; dd += 2) {
            int col = h * 32 + dd;
            float o0 = p0 * vb[col]     + p1 * vb[DM + col]     + p2 * vb[2*DM + col];
            float o1 = p0 * vb[col + 1] + p1 * vb[DM + col + 1] + p2 * vb[2*DM + col + 1];
            half2v hv = { (_Float16)o0, (_Float16)o1 };
            *(half2v*)(oA + row * T_LDOA + col) = hv;
        }
    }
    __syncthreads();

    // Phase 2: FC GEMM  x[48 x 256] = o[48 x 256] @ fc_w[256 x 256]
    float8 acc[3][2] = {};
    for (int ks = 0; ks < 8; ++ks) {
        const int k0 = ks * 32;
        // fc_w rows k0..k0+31 -> bT[n][k] via k-pair half2 stores (no guards, 256|)
        #pragma unroll
        for (int j = 0; j < 4; ++j) {
            int item = j * 256 + tid;               // 1024 items
            int kk   = (item >> 6) * 2;             // 0,2,..,30
            int n4   = (item & 63) * 4;             // 0..252
            const float* base = fcw + (size_t)(k0 + kk) * DM + n4;
            float4v r0 = *(const float4v*)base;
            float4v r1 = *(const float4v*)(base + DM);
            #pragma unroll
            for (int i = 0; i < 4; ++i) {
                half2v h = { (_Float16)r0[i], (_Float16)r1[i] };
                *(half2v*)(bT + (n4 + i) * T_LDBT + kk) = h;
            }
        }
        __syncthreads();
        half16 bf[2];
        #pragma unroll
        for (int nt = 0; nt < 2; ++nt)
            bf[nt] = load_b_frag(bT, lane, wave * 32 + nt * 16, T_LDBT);
        #pragma unroll
        for (int mt = 0; mt < 3; ++mt) {
            half16 af = load_a_frag(oA + k0, lane, mt * 16, T_LDOA);
            #pragma unroll
            for (int nt = 0; nt < 2; ++nt)
                acc[mt][nt] = wmma32(af, bf[nt], acc[mt][nt]);
        }
        __syncthreads();
    }

    // Phase 3: x = fc(o) + fc_b + residual(v) -> xs (overwrites oA/bT)
    const int g  = lane >> 4;
    const int nl = lane & 15;
    #pragma unroll
    for (int mt = 0; mt < 3; ++mt) {
        #pragma unroll
        for (int nt = 0; nt < 2; ++nt) {
            int col = wave * 32 + nt * 16 + nl;
            #pragma unroll
            for (int i = 0; i < 8; ++i) {
                int row = mt * 16 + g * 8 + i;
                int si = row / 3, mi = row % 3;
                float val = acc[mt][nt][i] + fcb[col]
                          + vbuf[(size_t)(b0 + si) * NQKV + mi * DM + col];
                xs[row * T_LDX + col] = val;
            }
        }
    }
    __syncthreads();

    // Phase 4: LayerNorm per row (wave32 shuffle reduction); 8 waves * 6 rows
    for (int q = 0; q < 6; ++q) {
        int row = wave * 6 + q;
        float s = 0.0f, s2 = 0.0f, vals[8];
        #pragma unroll
        for (int j = 0; j < 8; ++j) {
            int col = lane + j * 32;
            float v = xs[row * T_LDX + col];
            vals[j] = v; s += v; s2 += v * v;
        }
        #pragma unroll
        for (int off = 16; off > 0; off >>= 1) {
            s  += __shfl_xor(s,  off, 32);
            s2 += __shfl_xor(s2, off, 32);
        }
        float mu   = s * (1.0f / 256.0f);
        float var  = s2 * (1.0f / 256.0f) - mu * mu;
        float rstd = rsqrtf(var + 1e-6f);
        #pragma unroll
        for (int j = 0; j < 8; ++j) {
            int col = lane + j * 32;
            xs[row * T_LDX + col] = (vals[j] - mu) * rstd * lng[col] + lnb[col];
        }
    }
    __syncthreads();

    // Phase 5: output head (deterministic, no atomics)
    if (tid < T_SPB * 5) {
        int si = tid / 5, cls = tid % 5;
        float a = outb[cls];
        for (int mi = 0; mi < 3; ++mi)
            for (int col = 0; col < DM; ++col)
                a += xs[(si * 3 + mi) * T_LDX + col] * outw[(size_t)(mi * DM + col) * 5 + cls];
        out[(size_t)(b0 + si) * 5 + cls] = a;
    }
}

// ---------------- host launcher -------------------------------------------------
extern "C" void kernel_launch(void* const* d_in, const int* in_sizes, int n_in,
                              void* d_out, int out_size, void* d_ws, size_t ws_size,
                              hipStream_t stream) {
    ProjParams pp;
    pp.X = (const float*)d_in[0];
    for (int seg = 0; seg < 3; ++seg)
        for (int m = 0; m < 3; ++m) {
            pp.W[seg * 3 + m]  = (const float*)d_in[1 + seg * 6 + m * 2];
            pp.Bi[seg * 3 + m] = (const float*)d_in[2 + seg * 6 + m * 2];
        }
    const float* fcw  = (const float*)d_in[19];
    const float* fcb  = (const float*)d_in[20];
    const float* lng  = (const float*)d_in[21];
    const float* lnb  = (const float*)d_in[22];
    const float* outw = (const float*)d_in[23];
    const float* outb = (const float*)d_in[24];

    float* qbuf = (float*)d_ws;
    float* kbuf = qbuf + (size_t)BS * NQKV;
    float* vbuf = kbuf + (size_t)BS * NQKV;
    float* attn = vbuf + (size_t)BS * NQKV;
    unsigned* gmin = (unsigned*)(attn + (size_t)BS * 72);
    pp.outbase = qbuf;

    init_gmin_kernel<<<1, 1, 0, stream>>>(gmin);
    proj_gemm<<<dim3(BS / P_BM, NQKV / P_BN, 3), 256, 0, stream>>>(pp);
    attn_scores<<<(BS * 72) / 256, 256, 0, stream>>>(qbuf, kbuf, attn, gmin);
    fused_tail<<<BS / T_SPB, 256, 0, stream>>>(attn, vbuf, gmin, fcw, fcb, lng, lnb,
                                               outw, outb, (float*)d_out);
}
